// RNNModelWithConditioning_28174985462582
// MI455X (gfx1250) — matrix operand order repaired and checked by reading, not compiled, not run on hardware
//
#include <hip/hip_runtime.h>
#include <hip/hip_bf16.h>
#include <math.h>
#include <stdint.h>

// ---------------- model dims ----------------
#define NB    256          // batch
#define NT    256          // time steps
#define NIN   81           // n_input
#define NINP  96           // n_input padded to mult of 32 (K of in1/in2 GEMMs)
#define NH    1024         // n_hidden
#define NCH   256          // n_cond_hidden
#define NCOND 128          // n_cond
#define RNNIN 1984         // 64 filters * 31
#define K1    (RNNIN + NCH)  // 2240: GRU1 input width
#define K23   (NH + NCH)     // 1280: GRU2/3 input width
#define G3    (3 * NH)       // 3072: gate width
#define NOUT  81
#define NOUTP 128          // final_w rows padded to mult of 64

typedef __attribute__((ext_vector_type(16))) __bf16 v16bf;
typedef __attribute__((ext_vector_type(8)))  float  v8f;

// gfx1250 async global->LDS copy (ASYNCcnt tracked), per-lane 16B
__device__ __forceinline__ void async_copy_b128(const void* gp, void* lp)
{
    unsigned lds = (unsigned)(uintptr_t)lp;                 // low 32b = LDS offset
    unsigned long long ga = (unsigned long long)(uintptr_t)gp;
    asm volatile("global_load_async_to_lds_b128 %0, %1, off"
                 :: "v"(lds), "v"(ga) : "memory");
}
__device__ __forceinline__ void wait_async0()
{
    asm volatile("s_wait_asynccnt 0x0" ::: "memory");
}

// =====================================================================
// bf16 WMMA GEMM:  C[M,N] = A[M,K] * W[N,K]^T (+bias) (+=C)
// Strict contract: M%64==0, K%32==0, lda%8==0, ldw%8==0, W padded to
// ceil(N/64)*64 zero rows. 8 waves, WG tile 64x64, wave tile 16x32.
// Double-buffered LDS staging via global_load_async_to_lds_b128.
// =====================================================================
__global__ __launch_bounds__(256)
void wmma_gemm_bf16(const __bf16* __restrict__ A, int lda,
                    const __bf16* __restrict__ W, int ldw,
                    const float* __restrict__ bias,
                    float* __restrict__ C, int ldc,
                    __bf16* __restrict__ Cb, int ldcb,
                    int N, int K, int accumulate)
{
    // row padded to 40 bf16 (80B = 20 banks) -> conflict-free b128 frag loads
    __shared__ __align__(16) __bf16 As[2][64][40];
    __shared__ __align__(16) __bf16 Ws[2][64][40];

    const int tid  = threadIdx.x;
    const int m0   = blockIdx.x * 64;
    const int n0   = blockIdx.y * 64;
    const int wv   = tid >> 5;
    const int wr   = wv >> 1;        // wave row block (0..3) -> 16 rows
    const int wc   = wv & 1;         // wave col block (0..1) -> 32 cols
    const int lane = tid & 31;
    const int half = lane >> 4;
    const int l15  = lane & 15;

    const int sr = tid >> 2;         // staging row 0..63
    const int sc = (tid & 3) * 8;    // staging col 0,8,16,24

    const __bf16* aRow = A + (size_t)(m0 + sr) * lda + sc;
    const __bf16* wRow = W + (size_t)(n0 + sr) * ldw + sc;

    v8f acc0 = {}; v8f acc1 = {};

    // prime buffer 0
    async_copy_b128(aRow, &As[0][sr][sc]);
    async_copy_b128(wRow, &Ws[0][sr][sc]);
    wait_async0();
    __syncthreads();

    for (int k0 = 0; k0 < K; k0 += 32) {
        const int cur = (k0 >> 5) & 1;
        const int nxt = cur ^ 1;
        const bool more = (k0 + 32) < K;
        if (more) {  // overlap next tile's global->LDS with this tile's WMMAs
            async_copy_b128(aRow + (k0 + 32), &As[nxt][sr][sc]);
            async_copy_b128(wRow + (k0 + 32), &Ws[nxt][sr][sc]);
        }

        // fragments: ISA 16-bit layouts, 2 x ds_load_b128 each
        union Frag { int4 q[2]; v16bf v; } af, bf0, bf1;
        const int ar = wr * 16 + l15;                          // A: M=l15
        af.q[0] = *(const int4*)&As[cur][ar][half * 8];        // K half*8..+7
        af.q[1] = *(const int4*)&As[cur][ar][16 + half * 8];   // K 16+half*8..+7
        const int bn0 = wc * 32 + l15;
        bf0.q[0] = *(const int4*)&Ws[cur][bn0][half * 8];
        bf0.q[1] = *(const int4*)&Ws[cur][bn0][16 + half * 8];
        const int bn1 = wc * 32 + 16 + l15;
        bf1.q[0] = *(const int4*)&Ws[cur][bn1][half * 8];
        bf1.q[1] = *(const int4*)&Ws[cur][bn1][16 + half * 8];

        acc0 = __builtin_amdgcn_wmma_f32_16x16x32_bf16(false, af.v, false, bf0.v,
                                                       (short)0, acc0, false, false);
        acc1 = __builtin_amdgcn_wmma_f32_16x16x32_bf16(false, af.v, false, bf1.v,
                                                       (short)0, acc1, false, false);

        if (more) wait_async0();
        __syncthreads();
    }

    // store C (16x16 f32 layout: VGPR r -> row r + 8*half, col = l15)
    for (int c = 0; c < 2; ++c) {
        const v8f a = c ? acc1 : acc0;
        const int col = n0 + wc * 32 + c * 16 + l15;
        if (col >= N) continue;
        const float bs = bias ? bias[col] : 0.0f;
        for (int r = 0; r < 8; ++r) {
            const int row = m0 + wr * 16 + half * 8 + r;
            float v = a[r] + bs;
            if (accumulate) v += C[(size_t)row * ldc + col];
            C[(size_t)row * ldc + col] = v;
            if (Cb) Cb[(size_t)row * ldcb + col] = (__bf16)v;
        }
    }
}

// =====================================================================
// Elementwise helpers
// =====================================================================
__global__ void k_f2b(const float* __restrict__ in, __bf16* __restrict__ out, int n)
{
    int i = blockIdx.x * 256 + threadIdx.x;
    if (i < n) out[i] = (__bf16)in[i];
}

// convert + zero-pad: dst[r][c] (ldd cols, rowsPad rows) from src[rows][cols]
__global__ void k_f2b_pad(const float* __restrict__ src, __bf16* __restrict__ dst,
                          int rows, int cols, int ldd, int n)
{
    int i = blockIdx.x * 256 + threadIdx.x;
    if (i >= n) return;
    int r = i / ldd, c = i % ldd;
    dst[i] = (r < rows && c < cols) ? (__bf16)src[(size_t)r * cols + c] : (__bf16)0.0f;
}

__global__ void k_copy_state(const float* __restrict__ in, float* __restrict__ of,
                             __bf16* __restrict__ ob)
{
    int i = blockIdx.x * 256 + threadIdx.x;  // n = NB*NH exact
    float v = in[i];
    of[i] = v;
    ob[i] = (__bf16)v;
}

// GRU nonlinearity: h = (1-z)*tanh(xn + r*hn) + z*h ; emit bf16 mirror
__global__ void k_gru(const float* __restrict__ gx, const float* __restrict__ gh,
                      float* __restrict__ h, __bf16* __restrict__ hb)
{
    int i = blockIdx.x * 256 + threadIdx.x;  // n = NB*NH exact
    int b = i >> 10, c = i & 1023;
    size_t o = (size_t)b * G3 + c;
    float r = 1.0f / (1.0f + __expf(-(gx[o]           + gh[o])));
    float z = 1.0f / (1.0f + __expf(-(gx[o + NH]      + gh[o + NH])));
    float n = tanhf(gx[o + 2 * NH] + r * gh[o + 2 * NH]);
    float hv = (1.0f - z) * n + z * h[i];
    h[i]  = hv;
    hb[i] = (__bf16)hv;
}

__global__ void k_combine2(const float* __restrict__ a, const float* __restrict__ b,
                           __bf16* __restrict__ out, int ldo)
{
    int i = blockIdx.x * 256 + threadIdx.x;  // n = NB*NH exact
    int r = i >> 10, c = i & 1023;
    out[(size_t)r * ldo + c] = (__bf16)(a[i] + b[i]);
}

__global__ void k_combine3(const float* __restrict__ a, const float* __restrict__ b,
                           const float* __restrict__ cc, __bf16* __restrict__ out, int ldo)
{
    int i = blockIdx.x * 256 + threadIdx.x;
    int r = i >> 10, c = i & 1023;
    out[(size_t)r * ldo + c] = (__bf16)(a[i] + b[i] + cc[i]);
}

// x0_t[b, f] = x[b, 0, f, t] -> bf16 [256 x 96] (cols 81..95 zero)
__global__ void k_x0(const float* __restrict__ x, __bf16* __restrict__ out, int t)
{
    int i = blockIdx.x * 256 + threadIdx.x;  // n = NB*NINP = 24576 exact
    int b = i / NINP, f = i % NINP;
    out[i] = (f < NIN) ? (__bf16)x[(size_t)b * NIN * NT + (size_t)f * NT + t]
                       : (__bf16)0.0f;
}

// Conv front-end for one timestep -> GRU1 concat input cols [0,1984)
__global__ void k_conv(const float* __restrict__ x, const float* __restrict__ w,
                       const float* __restrict__ cb, __bf16* __restrict__ xcat, int t)
{
    int i = blockIdx.x * 256 + threadIdx.x;  // n = NB*RNNIN = 507904 exact
    int b  = i / RNNIN;
    int fo = i % RNNIN;        // feature index = f*31 + ho
    int f  = fo / 31;
    int ho = fo % 31;
    float acc = cb[f];
    const float* xb = x + (size_t)b * NIN * NT;
    const float* wf = w + f * (21 * 11);
    for (int kf = 0; kf < 21; ++kf) {
        const float* xr = xb + (size_t)(ho * 2 + kf) * NT;
        const float* wr = wf + kf * 11;
        for (int kt = 0; kt < 11; ++kt) {
            int tt = t + kt - 5;
            if (tt >= 0 && tt < NT) acc += xr[tt] * wr[kt];
        }
    }
    xcat[(size_t)b * K1 + fo] = (__bf16)acc;
}

// =====================================================================
// Host-side orchestration
// =====================================================================
extern "C" void kernel_launch(void* const* d_in, const int* in_sizes, int n_in,
                              void* d_out, int out_size, void* d_ws, size_t ws_size,
                              hipStream_t stream)
{
    const float* x      = (const float*)d_in[0];
    const float* cond   = (const float*)d_in[1];
    const float* h1in   = (const float*)d_in[2];
    const float* h2in   = (const float*)d_in[3];
    const float* h3in   = (const float*)d_in[4];
    const float* conv_w = (const float*)d_in[5];
    const float* conv_b = (const float*)d_in[6];
    const float* Wih0 = (const float*)d_in[7],  *Whh0 = (const float*)d_in[8];
    const float* Wih1 = (const float*)d_in[9],  *Whh1 = (const float*)d_in[10];
    const float* Wih2 = (const float*)d_in[11], *Whh2 = (const float*)d_in[12];
    const float* in1_w = (const float*)d_in[13], *in1_b = (const float*)d_in[14];
    const float* in2_w = (const float*)d_in[15], *in2_b = (const float*)d_in[16];
    const float* cw0 = (const float*)d_in[17], *cb0 = (const float*)d_in[18];
    const float* cw1 = (const float*)d_in[19], *cb1 = (const float*)d_in[20];
    const float* cw2 = (const float*)d_in[21], *cb2 = (const float*)d_in[22];
    const float* ht0_w = (const float*)d_in[23], *ht0_b = (const float*)d_in[24];
    const float* ht1_w = (const float*)d_in[25], *ht1_b = (const float*)d_in[26];
    const float* ht2_w = (const float*)d_in[27], *ht2_b = (const float*)d_in[28];
    const float* out0_w = (const float*)d_in[29], *out0_b = (const float*)d_in[30];
    const float* out1_w = (const float*)d_in[31], *out1_b = (const float*)d_in[32];
    const float* out2_w = (const float*)d_in[33], *out2_b = (const float*)d_in[34];
    const float* fin_w = (const float*)d_in[35], *fin_b = (const float*)d_in[36];
    float* out = (float*)d_out;

    // ---- workspace carve-out (bump allocator, 256B aligned) ----
    char* ws = (char*)d_ws;
    size_t off = 0;
    auto alloc = [&](size_t bytes) -> void* {
        void* p = ws + off;
        off = (off + bytes + 255) & ~(size_t)255;
        return p;
    };
    __bf16* WIH0b = (__bf16*)alloc((size_t)G3 * K1 * 2);
    __bf16* WHH0b = (__bf16*)alloc((size_t)G3 * NH * 2);
    __bf16* WIH1b = (__bf16*)alloc((size_t)G3 * K23 * 2);
    __bf16* WHH1b = (__bf16*)alloc((size_t)G3 * NH * 2);
    __bf16* WIH2b = (__bf16*)alloc((size_t)G3 * K23 * 2);
    __bf16* WHH2b = (__bf16*)alloc((size_t)G3 * NH * 2);
    __bf16* HT0b  = (__bf16*)alloc((size_t)NH * NH * 2);
    __bf16* HT1b  = (__bf16*)alloc((size_t)NH * NH * 2);
    __bf16* HT2b  = (__bf16*)alloc((size_t)NH * NH * 2);
    __bf16* OUT0b = (__bf16*)alloc((size_t)NH * NH * 2);
    __bf16* OUT1b = (__bf16*)alloc((size_t)NH * NH * 2);
    __bf16* OUT2b = (__bf16*)alloc((size_t)NH * NH * 2);
    __bf16* FINb  = (__bf16*)alloc((size_t)NOUTP * NH * 2);   // padded 128 rows
    __bf16* IN1b  = (__bf16*)alloc((size_t)NH * NINP * 2);    // padded K=96
    __bf16* IN2b  = (__bf16*)alloc((size_t)NH * NINP * 2);
    __bf16* CW0b  = (__bf16*)alloc((size_t)NCH * NCOND * 2);
    __bf16* CW1b  = (__bf16*)alloc((size_t)NCH * NCOND * 2);
    __bf16* CW2b  = (__bf16*)alloc((size_t)NCH * NCOND * 2);
    __bf16* CONDb = (__bf16*)alloc((size_t)NB * NCOND * 2);
    __bf16* XCAT1 = (__bf16*)alloc((size_t)NB * K1 * 2);
    __bf16* XCAT2 = (__bf16*)alloc((size_t)NB * K23 * 2);
    __bf16* XCAT3 = (__bf16*)alloc((size_t)NB * K23 * 2);
    float*  GX    = (float*)alloc((size_t)NB * G3 * 4);
    float*  GH    = (float*)alloc((size_t)NB * G3 * 4);
    float*  H1    = (float*)alloc((size_t)NB * NH * 4);
    float*  H2    = (float*)alloc((size_t)NB * NH * 4);
    float*  H3    = (float*)alloc((size_t)NB * NH * 4);
    __bf16* H1b   = (__bf16*)alloc((size_t)NB * NH * 2);
    __bf16* H2b   = (__bf16*)alloc((size_t)NB * NH * 2);
    __bf16* H3b   = (__bf16*)alloc((size_t)NB * NH * 2);
    float*  BUFA  = (float*)alloc((size_t)NB * NH * 4);   // h1_to_h2
    __bf16* BUFAb = (__bf16*)alloc((size_t)NB * NH * 2);
    float*  BUFB  = (float*)alloc((size_t)NB * NH * 4);   // h1_to_h3
    float*  BUFC  = (float*)alloc((size_t)NB * NH * 4);   // h2_to_h3
    float*  BUFX2 = (float*)alloc((size_t)NB * NH * 4);   // x2_t
    float*  BUFX3 = (float*)alloc((size_t)NB * NH * 4);   // x3_t
    float*  RACC  = (float*)alloc((size_t)NB * NH * 4);   // r1+r2+r3
    __bf16* RACCb = (__bf16*)alloc((size_t)NB * NH * 2);
    float*  CONDF = (float*)alloc((size_t)NB * NCH * 4);  // throwaway fp32 C
    __bf16* X0b   = (__bf16*)alloc((size_t)NB * NINP * 2);
    (void)ws_size; (void)in_sizes; (void)n_in; (void)out_size;

    auto cvt = [&](const float* src, __bf16* dst, size_t n) {
        k_f2b<<<dim3((unsigned)((n + 255) / 256)), dim3(256), 0, stream>>>(src, dst, (int)n);
    };
    auto cvtPad = [&](const float* src, __bf16* dst, int rows, int cols, int ldd, int rowsPad) {
        int n = rowsPad * ldd;
        k_f2b_pad<<<dim3((unsigned)((n + 255) / 256)), dim3(256), 0, stream>>>(
            src, dst, rows, cols, ldd, n);
    };
    auto gemm = [&](const __bf16* A, int lda, const __bf16* W, int ldw,
                    const float* bias, float* C, int ldc, __bf16* Cb, int ldcb,
                    int M, int N, int K, int acc) {
        dim3 g((unsigned)(M / 64), (unsigned)((N + 63) / 64));
        wmma_gemm_bf16<<<g, dim3(256), 0, stream>>>(A, lda, W, ldw, bias,
                                                    C, ldc, Cb, ldcb, N, K, acc);
    };

    // ---------------- setup: weight/cond conversion ----------------
    cvt(Wih0, WIH0b, (size_t)G3 * K1);
    cvt(Whh0, WHH0b, (size_t)G3 * NH);
    cvt(Wih1, WIH1b, (size_t)G3 * K23);
    cvt(Whh1, WHH1b, (size_t)G3 * NH);
    cvt(Wih2, WIH2b, (size_t)G3 * K23);
    cvt(Whh2, WHH2b, (size_t)G3 * NH);
    cvt(ht0_w, HT0b, (size_t)NH * NH);
    cvt(ht1_w, HT1b, (size_t)NH * NH);
    cvt(ht2_w, HT2b, (size_t)NH * NH);
    cvt(out0_w, OUT0b, (size_t)NH * NH);
    cvt(out1_w, OUT1b, (size_t)NH * NH);
    cvt(out2_w, OUT2b, (size_t)NH * NH);
    cvt(cw0, CW0b, (size_t)NCH * NCOND);
    cvt(cw1, CW1b, (size_t)NCH * NCOND);
    cvt(cw2, CW2b, (size_t)NCH * NCOND);
    cvt(cond, CONDb, (size_t)NB * NCOND);
    cvtPad(fin_w, FINb, NOUT, NH, NH, NOUTP);       // 81x1024 -> 128x1024
    cvtPad(in1_w, IN1b, NH, NIN, NINP, NH);         // 1024x81 -> 1024x96
    cvtPad(in2_w, IN2b, NH, NIN, NINP, NH);

    // hidden-state init (fp32 + bf16 mirror)
    k_copy_state<<<dim3(NB * NH / 256), dim3(256), 0, stream>>>(h1in, H1, H1b);
    k_copy_state<<<dim3(NB * NH / 256), dim3(256), 0, stream>>>(h2in, H2, H2b);
    k_copy_state<<<dim3(NB * NH / 256), dim3(256), 0, stream>>>(h3in, H3, H3b);

    // cond projections: bf16 mirror written straight into the concat tails
    gemm(CONDb, NCOND, CW0b, NCOND, cb0, CONDF, NCH, XCAT1 + RNNIN, K1,  NB, NCH, NCOND, 0);
    gemm(CONDb, NCOND, CW1b, NCOND, cb1, CONDF, NCH, XCAT2 + NH,    K23, NB, NCH, NCOND, 0);
    gemm(CONDb, NCOND, CW2b, NCOND, cb2, CONDF, NCH, XCAT3 + NH,    K23, NB, NCH, NCOND, 0);

    // ---------------- the scan ----------------
    const dim3 ew(256);
    for (int t = 0; t < NT; ++t) {
        k_conv<<<dim3(NB * RNNIN / 256), ew, 0, stream>>>(x, conv_w, conv_b, XCAT1, t);
        k_x0<<<dim3(NB * NINP / 256), ew, 0, stream>>>(x, X0b, t);

        // ---- layer 1 ----
        gemm(XCAT1, K1, WIH0b, K1, nullptr, GX, G3, nullptr, 0, NB, G3, K1, 0);
        gemm(H1b, NH, WHH0b, NH, nullptr, GH, G3, nullptr, 0, NB, G3, NH, 0);
        k_gru<<<dim3(NB * NH / 256), ew, 0, stream>>>(GX, GH, H1, H1b);

        // transitions + readout 1
        gemm(H1b, NH, HT0b, NH, ht0_b, BUFA, NH, BUFAb, NH, NB, NH, NH, 0);   // h1_to_h2
        gemm(H1b, NH, HT1b, NH, ht1_b, BUFB, NH, nullptr, 0, NB, NH, NH, 0);  // h1_to_h3
        gemm(H1b, NH, OUT0b, NH, out0_b, RACC, NH, nullptr, 0, NB, NH, NH, 0); // r1

        // ---- layer 2 input: x2_t + h1_to_h2 || cond2 ----
        gemm(X0b, NINP, IN1b, NINP, in1_b, BUFX2, NH, nullptr, 0, NB, NH, NINP, 0);
        k_combine2<<<dim3(NB * NH / 256), ew, 0, stream>>>(BUFX2, BUFA, XCAT2, K23);
        gemm(XCAT2, K23, WIH1b, K23, nullptr, GX, G3, nullptr, 0, NB, G3, K23, 0);
        gemm(H2b, NH, WHH1b, NH, nullptr, GH, G3, nullptr, 0, NB, G3, NH, 0);
        k_gru<<<dim3(NB * NH / 256), ew, 0, stream>>>(GX, GH, H2, H2b);

        // h2_to_h3 (reference quirk: from h1_to_h2) + readout 2
        gemm(BUFAb, NH, HT2b, NH, ht2_b, BUFC, NH, nullptr, 0, NB, NH, NH, 0);
        gemm(H2b, NH, OUT1b, NH, out1_b, RACC, NH, nullptr, 0, NB, NH, NH, 1); // r2 +=

        // ---- layer 3 input: x3_t + h1_to_h3 + h2_to_h3 || cond3 ----
        gemm(X0b, NINP, IN2b, NINP, in2_b, BUFX3, NH, nullptr, 0, NB, NH, NINP, 0);
        k_combine3<<<dim3(NB * NH / 256), ew, 0, stream>>>(BUFX3, BUFB, BUFC, XCAT3, K23);
        gemm(XCAT3, K23, WIH2b, K23, nullptr, GX, G3, nullptr, 0, NB, G3, K23, 0);
        gemm(H3b, NH, WHH2b, NH, nullptr, GH, G3, nullptr, 0, NB, G3, NH, 0);
        k_gru<<<dim3(NB * NH / 256), ew, 0, stream>>>(GX, GH, H3, H3b);

        // readout 3 (+= and emit bf16 mirror for final projection)
        gemm(H3b, NH, OUT2b, NH, out2_b, RACC, NH, RACCb, NH, NB, NH, NH, 1);

        // final projection straight into d_out[t]  (N=81, W padded to 128 rows)
        gemm(RACCb, NH, FINb, NH, fin_b,
             out + (size_t)t * NB * NOUT, NOUT, nullptr, 0, NB, NOUT, NH, 0);
    }
}
